// MultiHeadAttention_51273319580046
// MI455X (gfx1250) — matrix operand order repaired and checked by reading, not compiled, not run on hardware
//
#include <hip/hip_runtime.h>

#define S_LEN 2048
#define DM 1024
#define NH 16
#define DH 64
#define NB 4
#define QKV_ELEMS ((size_t)NB * NH * S_LEN * DH)   // 8,388,608 per tensor

typedef __bf16 bf16_t;
typedef __bf16 bf16x16 __attribute__((ext_vector_type(16)));
typedef float f32x8 __attribute__((ext_vector_type(8)));

struct alignas(16) B128v { unsigned int x, y, z, w; };
struct alignas(16) F4 { float x, y, z, w; };
struct alignas(8)  H4 { bf16_t a, b, c, d; };

union Frag16 {
    bf16x16 v;
    bf16_t  e[16];
    B128v   b[2];
};

static __device__ __forceinline__ void cvt4(bf16_t* d, F4 f) {
    d[0] = (bf16_t)f.x; d[1] = (bf16_t)f.y; d[2] = (bf16_t)f.z; d[3] = (bf16_t)f.w;
}

static __device__ __forceinline__ f32x8 wmma_bf16(const Frag16& a, const Frag16& b, f32x8 c) {
    // (neg_a, A, neg_b, B, c_mod, C, reuse_a, reuse_b)
    return __builtin_amdgcn_wmma_f32_16x16x32_bf16(false, a.v, false, b.v, (short)0, c, false, false);
}

// ---------------------------------------------------------------------------
// Kernel 0: one-shot fp32 -> bf16 conversion of the 4 weight matrices.
// ---------------------------------------------------------------------------
__global__ __launch_bounds__(256) void wconv_kernel(
    const float* __restrict__ Wq, const float* __restrict__ Wk,
    const float* __restrict__ Wv, const float* __restrict__ Wo,
    bf16_t* __restrict__ wbf)
{
    const int z = blockIdx.z;
    const float* W = (z == 0) ? Wq : (z == 1) ? Wk : (z == 2) ? Wv : Wo;
    bf16_t* o = wbf + (size_t)z * DM * DM;
    const size_t idx = ((size_t)blockIdx.x * 256 + threadIdx.x) * 4;
    F4 f = *(const F4*)(W + idx);
    H4 h;
    h.a = (bf16_t)f.x; h.b = (bf16_t)f.y; h.c = (bf16_t)f.z; h.d = (bf16_t)f.w;
    *(H4*)(o + idx) = h;
}

// ---------------------------------------------------------------------------
// Kernel 1: fused QKV projection.  Y = X @ W^T + b, written split-head bf16.
// block = 128 thr (4 waves), tile 64x64 per block, wave = 16 rows x 64 cols.
// ---------------------------------------------------------------------------
__global__ __launch_bounds__(128) void qkv_proj_kernel(
    const float* __restrict__ q_in, const float* __restrict__ k_in, const float* __restrict__ v_in,
    const bf16_t* __restrict__ wbf,
    const float* __restrict__ bq, const float* __restrict__ bk, const float* __restrict__ bv,
    bf16_t* __restrict__ ws)
{
    const int z = blockIdx.z;
    const float*  X    = (z == 0) ? q_in : (z == 1) ? k_in : v_in;
    const bf16_t* W    = wbf + (size_t)z * DM * DM;
    const float*  bias = (z == 0) ? bq : (z == 1) ? bk : bv;
    bf16_t* out = ws + (size_t)z * QKV_ELEMS;

    const int lane = threadIdx.x & 31;
    const int wave = threadIdx.x >> 5;
    const int kG   = lane >> 4;        // K-half selector
    const int l16  = lane & 15;        // A-row / B-col within tile

    const int mBase = blockIdx.x * 64 + wave * 16;
    const int nBase = blockIdx.y * 64;

    const f32x8 zf = {0.f, 0.f, 0.f, 0.f, 0.f, 0.f, 0.f, 0.f};
    f32x8 acc[4] = {zf, zf, zf, zf};

    const float* xrow = X + (size_t)(mBase + l16) * DM;

    for (int kb = 0; kb < DM; kb += 32) {
        Frag16 a;
        {
            const float* p0 = xrow + kb + kG * 8;
            const float* p1 = xrow + kb + 16 + kG * 8;
            cvt4(&a.e[0],  *(const F4*)(p0));
            cvt4(&a.e[4],  *(const F4*)(p0 + 4));
            cvt4(&a.e[8],  *(const F4*)(p1));
            cvt4(&a.e[12], *(const F4*)(p1 + 4));
        }
#pragma unroll
        for (int nt = 0; nt < 4; ++nt) {
            Frag16 bf;
            const bf16_t* wp = W + (size_t)(nBase + nt * 16 + l16) * DM + kb;
            bf.b[0] = *(const B128v*)(wp + kG * 8);
            bf.b[1] = *(const B128v*)(wp + 16 + kG * 8);
            acc[nt] = wmma_bf16(a, bf, acc[nt]);
        }
    }

#pragma unroll
    for (int nt = 0; nt < 4; ++nt) {
        const int n = nBase + nt * 16 + l16;
        const float bval = bias[n];
        const int h = n >> 6, d = n & 63;
#pragma unroll
        for (int r = 0; r < 8; ++r) {
            const int m = mBase + kG * 8 + r;
            const int bb = m >> 11, s = m & (S_LEN - 1);
            out[((size_t)(bb * NH + h) * S_LEN + s) * DH + d] = (bf16_t)(acc[nt][r] + bval);
        }
    }
}

// ---------------------------------------------------------------------------
// Kernel 2: flash attention per (b,h).  block = 256 thr (8 waves), each wave
// owns 16 query rows; KV streamed in chunks of 32 with online softmax.
// ---------------------------------------------------------------------------
__global__ __launch_bounds__(256) void attention_kernel(
    const bf16_t* __restrict__ ws_qkv,
    const float* __restrict__ mask, const int* __restrict__ pad,
    bf16_t* __restrict__ ctx_ws)
{
    __shared__ bf16_t lds_p[8][16 * 32];           // per-wave P tile (fragment transpose)
    __shared__ unsigned short lds_vT[DH * 32];     // V chunk transposed [d][kpos]

    const int bh = blockIdx.x;
    const int b  = bh >> 4;
    const int hh = bh & 15;
    const int lane = threadIdx.x & 31;
    const int wave = threadIdx.x >> 5;
    const int kG   = lane >> 4;
    const int col  = lane & 15;
    const int row8 = kG * 8;
    const int qBase = blockIdx.y * 128 + wave * 16;

    const bf16_t* qp = ws_qkv + (size_t)bh * S_LEN * DH;
    const bf16_t* kp = ws_qkv + QKV_ELEMS + (size_t)bh * S_LEN * DH;
    const bf16_t* vp = ws_qkv + 2 * QKV_ELEMS + (size_t)bh * S_LEN * DH;

    // Q fragments (16 rows x 64 d = two K=32 fragments), loaded once
    Frag16 qf[2];
#pragma unroll
    for (int s2 = 0; s2 < 2; ++s2) {
        const bf16_t* p = qp + (size_t)(qBase + col) * DH + s2 * 32;
        qf[s2].b[0] = *(const B128v*)(p + kG * 8);
        qf[s2].b[1] = *(const B128v*)(p + 16 + kG * 8);
    }

    const f32x8 zf = {0.f, 0.f, 0.f, 0.f, 0.f, 0.f, 0.f, 0.f};
    f32x8 ctx[4] = {zf, zf, zf, zf};
    float m_r[8], l_r[8];
#pragma unroll
    for (int r = 0; r < 8; ++r) { m_r[r] = -1e30f; l_r[r] = 0.f; }

    for (int kb = 0; kb < S_LEN; kb += 32) {
        __syncthreads();                       // protect lds_vT WAR across waves
        // stage V chunk transposed: lds_vT[d*32 + kpos] = V[kb+kpos][d]
#pragma unroll
        for (int i = 0; i < 4; ++i) {
            const int flat = threadIdx.x + 256 * i;  // 0..1023 pairs
            const int kpos = flat >> 5;
            const int dp   = flat & 31;
            const unsigned int u = *(const unsigned int*)(vp + (size_t)(kb + kpos) * DH + dp * 2);
            lds_vT[(dp * 2) * 32 + kpos]     = (unsigned short)(u & 0xffffu);
            lds_vT[(dp * 2 + 1) * 32 + kpos] = (unsigned short)(u >> 16);
        }
        __syncthreads();

        // scores = Q x K^T for two 16-col tiles
        f32x8 sc0 = zf, sc1 = zf;
#pragma unroll
        for (int s2 = 0; s2 < 2; ++s2) {
            Frag16 kf;
            const bf16_t* p = kp + (size_t)(kb + col) * DH + s2 * 32;
            kf.b[0] = *(const B128v*)(p + kG * 8);
            kf.b[1] = *(const B128v*)(p + 16 + kG * 8);
            sc0 = wmma_bf16(qf[s2], kf, sc0);
        }
#pragma unroll
        for (int s2 = 0; s2 < 2; ++s2) {
            Frag16 kf;
            const bf16_t* p = kp + (size_t)(kb + 16 + col) * DH + s2 * 32;
            kf.b[0] = *(const B128v*)(p + kG * 8);
            kf.b[1] = *(const B128v*)(p + 16 + kG * 8);
            sc1 = wmma_bf16(qf[s2], kf, sc1);
        }

        const float pad0 = (pad[b * S_LEN + kb + col] == 1) ? -1e30f : 0.f;
        const float pad1 = (pad[b * S_LEN + kb + 16 + col] == 1) ? -1e30f : 0.f;

        float sv0[8], sv1[8], mnew[8];
#pragma unroll
        for (int r = 0; r < 8; ++r) {
            const int qr = qBase + row8 + r;
            sv0[r] = sc0[r] * 0.125f + mask[(size_t)qr * S_LEN + kb + col] + pad0;
            sv1[r] = sc1[r] * 0.125f + mask[(size_t)qr * S_LEN + kb + 16 + col] + pad1;
            float v = fmaxf(sv0[r], sv1[r]);
            v = fmaxf(v, __shfl_xor(v, 1, 32));
            v = fmaxf(v, __shfl_xor(v, 2, 32));
            v = fmaxf(v, __shfl_xor(v, 4, 32));
            v = fmaxf(v, __shfl_xor(v, 8, 32));
            mnew[r] = fmaxf(m_r[r], v);
        }
#pragma unroll
        for (int r = 0; r < 8; ++r) {
            const float alpha = __expf(m_r[r] - mnew[r]);
            m_r[r] = mnew[r];
            l_r[r] *= alpha;
            ctx[0][r] *= alpha; ctx[1][r] *= alpha;
            ctx[2][r] *= alpha; ctx[3][r] *= alpha;
            const float p0 = __expf(sv0[r] - mnew[r]);
            const float p1 = __expf(sv1[r] - mnew[r]);
            lds_p[wave][(row8 + r) * 32 + col]      = (bf16_t)p0;
            lds_p[wave][(row8 + r) * 32 + 16 + col] = (bf16_t)p1;
            float s = p0 + p1;
            s += __shfl_xor(s, 1, 32);
            s += __shfl_xor(s, 2, 32);
            s += __shfl_xor(s, 4, 32);
            s += __shfl_xor(s, 8, 32);
            l_r[r] += s;
        }

        // P fragment (A-layout) back from per-wave LDS; DS in-order within wave
        Frag16 pf;
        {
            const bf16_t* p = &lds_p[wave][col * 32];
            pf.b[0] = *(const B128v*)(p + kG * 8);
            pf.b[1] = *(const B128v*)(p + 16 + kG * 8);
        }
        // context += P x V
#pragma unroll
        for (int dt = 0; dt < 4; ++dt) {
            Frag16 vf;
            const unsigned short* p = &lds_vT[(dt * 16 + col) * 32];
            vf.b[0] = *(const B128v*)(p + kG * 8);
            vf.b[1] = *(const B128v*)(p + 16 + kG * 8);
            ctx[dt] = wmma_bf16(pf, vf, ctx[dt]);
        }
    }

    // normalize and write context merged back to [B, S, D] bf16
#pragma unroll
    for (int r = 0; r < 8; ++r) {
        const float inv = 1.0f / l_r[r];
        const int qr = qBase + row8 + r;
        const size_t base = ((size_t)(b * S_LEN + qr)) * DM + hh * DH;
#pragma unroll
        for (int dt = 0; dt < 4; ++dt)
            ctx_ws[base + dt * 16 + col] = (bf16_t)(ctx[dt][r] * inv);
    }
}

// ---------------------------------------------------------------------------
// Kernel 3: output projection  out = ctx @ Wo^T + bo  (fp32 result)
// ---------------------------------------------------------------------------
__global__ __launch_bounds__(128) void out_proj_kernel(
    const bf16_t* __restrict__ ctx_ws,
    const bf16_t* __restrict__ Wo_bf, const float* __restrict__ bo,
    float* __restrict__ out)
{
    const int lane = threadIdx.x & 31;
    const int wave = threadIdx.x >> 5;
    const int kG   = lane >> 4;
    const int l16  = lane & 15;

    const int mBase = blockIdx.x * 64 + wave * 16;
    const int nBase = blockIdx.y * 64;

    const f32x8 zf = {0.f, 0.f, 0.f, 0.f, 0.f, 0.f, 0.f, 0.f};
    f32x8 acc[4] = {zf, zf, zf, zf};

    const bf16_t* arow = ctx_ws + (size_t)(mBase + l16) * DM;

    for (int kb = 0; kb < DM; kb += 32) {
        Frag16 a;
        a.b[0] = *(const B128v*)(arow + kb + kG * 8);
        a.b[1] = *(const B128v*)(arow + kb + 16 + kG * 8);
#pragma unroll
        for (int nt = 0; nt < 4; ++nt) {
            Frag16 bf;
            const bf16_t* wp = Wo_bf + (size_t)(nBase + nt * 16 + l16) * DM + kb;
            bf.b[0] = *(const B128v*)(wp + kG * 8);
            bf.b[1] = *(const B128v*)(wp + 16 + kG * 8);
            acc[nt] = wmma_bf16(a, bf, acc[nt]);
        }
    }

#pragma unroll
    for (int nt = 0; nt < 4; ++nt) {
        const int n = nBase + nt * 16 + l16;
        const float bval = bo[n];
#pragma unroll
        for (int r = 0; r < 8; ++r)
            out[(size_t)(mBase + kG * 8 + r) * DM + n] = acc[nt][r] + bval;
    }
}

// ---------------------------------------------------------------------------
extern "C" void kernel_launch(void* const* d_in, const int* in_sizes, int n_in,
                              void* d_out, int out_size, void* d_ws, size_t ws_size,
                              hipStream_t stream) {
    const float* query = (const float*)d_in[0];
    const float* key   = (const float*)d_in[1];
    const float* value = (const float*)d_in[2];
    const float* mask  = (const float*)d_in[3];
    const int*   pmask = (const int*)d_in[4];
    const float* Wq = (const float*)d_in[5];  const float* bq = (const float*)d_in[6];
    const float* Wk = (const float*)d_in[7];  const float* bk = (const float*)d_in[8];
    const float* Wv = (const float*)d_in[9];  const float* bv = (const float*)d_in[10];
    const float* Wo = (const float*)d_in[11]; const float* bo = (const float*)d_in[12];
    float* out = (float*)d_out;

    // ws layout (bf16 elems): q | k | v | ctx | Wq_bf | Wk_bf | Wv_bf | Wo_bf
    bf16_t* ws  = (bf16_t*)d_ws;
    bf16_t* ctx = ws + 3 * QKV_ELEMS;
    bf16_t* wbf = ws + 4 * QKV_ELEMS;

    wconv_kernel<<<dim3(DM * DM / 1024, 1, 4), 256, 0, stream>>>(
        Wq, Wk, Wv, Wo, wbf);
    qkv_proj_kernel<<<dim3(128, 16, 3), 128, 0, stream>>>(
        query, key, value, wbf, bq, bk, bv, ws);
    attention_kernel<<<dim3(NB * NH, S_LEN / 128), 256, 0, stream>>>(
        ws, mask, pmask, ctx);
    out_proj_kernel<<<dim3(128, 16), 128, 0, stream>>>(
        ctx, wbf + (size_t)3 * DM * DM, bo, out);
}
